// WindowAttention_69655779606749
// MI455X (gfx1250) — compile-verified, matrix operand order checked by reading, not run on hardware
//
#include <hip/hip_runtime.h>

typedef __attribute__((ext_vector_type(16))) _Float16 v16h;
typedef __attribute__((ext_vector_type(8)))  _Float16 v8h;
typedef __attribute__((ext_vector_type(4)))  _Float16 v4h;
typedef __attribute__((ext_vector_type(8)))  float    v8f;
typedef __attribute__((ext_vector_type(4)))  float    v4f;

#define WSZ 7
#define NH 4
#define NTOK 49          // 7*7
#define DIM 128
#define HD 32

// LDS strides (in halves / floats); f16 row strides are multiples of 16 bytes
#define XH_STRIDE 136    // f16 [64][128] + pad 8
#define QK_STRIDE 40     // f16 per-head [64][32] + pad 8
#define VT_STRIDE 72     // f16 per-head transposed [32][64] + pad 8
#define S_STRIDE  65     // f32 [64][64] + pad 1
#define P_STRIDE  72     // f16 [64][64] + pad 8

// ---------------------------------------------------------------------------
// WMMA helpers
// ---------------------------------------------------------------------------
__device__ __forceinline__ v8f wmma_f16(v16h a, v16h b, v8f c) {
  return __builtin_amdgcn_wmma_f32_16x16x32_f16(
      /*neg_a=*/false, a, /*neg_b=*/false, b,
      /*c_mod=*/(short)0, c, /*reuse_a=*/false, /*reuse_b=*/false);
}

// Load a 16x32 (MxK) A-fragment, or a 32x16 (KxN) B-fragment stored
// column-major (16 "rows" of 32 contiguous K values), from row-major storage
// with `stride_h` halves per row, K window starting at k0.
// CDNA5 layout: lanes 0-15 hold K = k0+[0..7] and k0+[16..23],
// lanes 16-31 hold K = k0+[8..15] and k0+[24..31].
__device__ __forceinline__ v16h frag_load(const _Float16* __restrict__ base,
                                          int lane, int stride_h, int k0) {
  int r  = lane & 15;
  int ko = k0 + ((lane & 16) >> 1);     // +8 halves for upper half-wave
  const _Float16* p = base + r * stride_h + ko;
  v8h lo = *(const v8h*)p;              // K .. K+7       (16B)
  v8h hi = *(const v8h*)(p + 16);       // K+16 .. K+23   (16B)
  v16h f;
#pragma unroll
  for (int i = 0; i < 8; ++i) { f[i] = lo[i]; f[i + 8] = hi[i]; }
  return f;
}

// ---------------------------------------------------------------------------
// Weight prep: fp32 -> f16, transposed to column-major so B-fragments are
// contiguous. wq: [384 cols][128 K], wp: [128 cols][128 K].
// ---------------------------------------------------------------------------
__global__ void prep_weights(const float* __restrict__ qkv_w,
                             const float* __restrict__ proj_w,
                             _Float16* __restrict__ wq,
                             _Float16* __restrict__ wp) {
  int e = blockIdx.x * blockDim.x + threadIdx.x;
  if (e < 384 * 128) {
    int n = e >> 7, k = e & 127;
    wq[n * 128 + k] = (_Float16)qkv_w[k * 384 + n];
  } else if (e < 384 * 128 + 128 * 128) {
    int e2 = e - 384 * 128;
    int n = e2 >> 7, k = e2 & 127;
    wp[n * 128 + k] = (_Float16)proj_w[k * 128 + n];
  }
}

// ---------------------------------------------------------------------------
// Main kernel: one workgroup per 7x7 window, 128 threads = 4 wave32.
// ---------------------------------------------------------------------------
struct alignas(16) Region0U {
  union {
    _Float16 Xh[64 * XH_STRIDE];        // stage 1-2: f16 input window
    float    S[NH * 64 * S_STRIDE];     // stage 3-4: f32 attention scores
    _Float16 Oh[64 * XH_STRIDE];        // stage 5-6: f16 attention output
  };
};
struct alignas(16) Region1U {
  union {
    struct {
      _Float16 Q[NH * 64 * QK_STRIDE];  // live through stage 3
      _Float16 K[NH * 64 * QK_STRIDE];
    } qk;
    _Float16 P[NH * 64 * P_STRIDE];     // stage 4-5: softmaxed probs
  };
};

__global__ __launch_bounds__(128) void swin_window_attention(
    const float* __restrict__ x,
    const float* __restrict__ qkv_b,
    const float* __restrict__ proj_b,
    const float* __restrict__ bias_table,
    const _Float16* __restrict__ wqkvT,   // [384][128] f16 col-major
    const _Float16* __restrict__ wprojT,  // [128][128] f16 col-major
    float* __restrict__ out) {
  __shared__ Region0U R0;
  __shared__ Region1U R1;
  __shared__ alignas(16) _Float16 Vt[NH * HD * VT_STRIDE];

  const int tid  = threadIdx.x;
  const int lane = tid & 31;
  const int wv   = tid >> 5;

  const int w  = blockIdx.x;       // 0..4095
  const int b  = w >> 6;           // batch
  const int wy = (w >> 3) & 7;     // window row
  const int wx = w & 7;            // window col
  const float* xwin = x + ((size_t)((b * 56 + wy * WSZ) * 56 + wx * WSZ)) * DIM;
  float* owin = out + ((size_t)((b * 56 + wy * WSZ) * 56 + wx * WSZ)) * DIM;

  // ---------------- Stage 1: stage window into LDS as f16 A matrix --------
  for (int idx = tid; idx < 64 * 32; idx += 128) {
    int row = idx >> 5, c4 = idx & 31;      // 32 float4 per row
    v4h hv = {};
    if (row < NTOK) {
      const float* src =
          xwin + (size_t)((row / WSZ) * 56 + (row % WSZ)) * DIM + c4 * 4;
      v4f f = *(const v4f*)src;
#pragma unroll
      for (int i = 0; i < 4; ++i) hv[i] = (_Float16)f[i];
    }
    *(v4h*)&R0.Xh[row * XH_STRIDE + c4 * 4] = hv;
  }
  __syncthreads();

  // ---------------- Stage 2: QKV GEMM [64,128] x [128,384] ----------------
  const float inv_sqrt_hd = 0.17677669529663687f;  // 32^-0.5 folded into Q
  for (int mt = 0; mt < 4; ++mt) {
    // Hoist the 4 K-step A-fragments of this M tile: reused by 6 N-tiles.
    v16h a[4];
#pragma unroll
    for (int kk = 0; kk < 4; ++kk)
      a[kk] = frag_load(&R0.Xh[mt * 16 * XH_STRIDE], lane, XH_STRIDE, kk * 32);
    for (int nt = wv; nt < 24; nt += 4) {
      v8f acc = {};
#pragma unroll
      for (int kk = 0; kk < 4; ++kk) {
        v16h bf = frag_load(wqkvT + nt * 16 * DIM, lane, DIM, kk * 32);
        acc = wmma_f16(a[kk], bf, acc);
      }
      // Epilogue: which/h/d derive from the tile index nt -> single branch.
      int n  = lane & 15;
      int mb = (lane & 16) >> 1;          // +8 rows for upper half-wave
      int which = nt >> 3;                // 0=q, 1=k, 2=v
      int h     = (nt >> 1) & 3;          // head
      int d     = ((nt & 1) << 4) + n;    // column within head
      float bias = qkv_b[nt * 16 + n];
      if (which == 2) {
        // V stored transposed: 8 rows of this lane contiguous -> one b128
        v8h vv;
#pragma unroll
        for (int j = 0; j < 8; ++j) vv[j] = (_Float16)(acc[j] + bias);
        *(v8h*)&Vt[(h * HD + d) * VT_STRIDE + mt * 16 + mb] = vv;
      } else {
        float scale = (which == 0) ? inv_sqrt_hd : 1.0f;
        _Float16* dst = (which == 0 ? R1.qk.Q : R1.qk.K) +
                        (h * 64 + mt * 16 + mb) * QK_STRIDE + d;
#pragma unroll
        for (int j = 0; j < 8; ++j)
          dst[j * QK_STRIDE] = (_Float16)((acc[j] + bias) * scale);
      }
    }
  }
  __syncthreads();

  // ---------------- Stage 3: scores S = Q K^T (one wave per head) ---------
  {
    int h = wv;
    const _Float16* Qh = &R1.qk.Q[h * 64 * QK_STRIDE];
    const _Float16* Kh = &R1.qk.K[h * 64 * QK_STRIDE];
    float* Sh = &R0.S[h * 64 * S_STRIDE];
    // Hoist all 4 K fragments (B operands) of this head.
    v16h kf[4];
#pragma unroll
    for (int nt = 0; nt < 4; ++nt)
      kf[nt] = frag_load(Kh + nt * 16 * QK_STRIDE, lane, QK_STRIDE, 0);
#pragma unroll
    for (int mt = 0; mt < 4; ++mt) {
      v16h a = frag_load(Qh + mt * 16 * QK_STRIDE, lane, QK_STRIDE, 0);
#pragma unroll
      for (int nt = 0; nt < 4; ++nt) {
        v8f acc = {};
        acc = wmma_f16(a, kf[nt], acc);  // K=32 in a single WMMA
        int n = lane & 15, mb = (lane & 16) >> 1;
#pragma unroll
        for (int j = 0; j < 8; ++j)
          Sh[(mt * 16 + mb + j) * S_STRIDE + nt * 16 + n] = acc[j];
      }
    }
  }
  __syncthreads();

  // ---------------- Stage 4: + relative bias, softmax, -> f16 P -----------
  for (int r = tid; r < NH * 64; r += 128) {
    int h = r >> 6, row = r & 63;
    _Float16* Prow = &R1.P[(h * 64 + row) * P_STRIDE];
    if (row < NTOK) {
      float* Srow = &R0.S[(h * 64 + row) * S_STRIDE];
      int qy = row / WSZ, qx = row % WSZ;
      // pass 1: add relative-position bias, track max (no div/mod in loop)
      float mx = -1e30f;
      {
        int c = 0;
        for (int ky = 0; ky < WSZ; ++ky) {
          int by = ((qy - ky + 6) * 13 + qx + 6) * NH + h;  // kx folded below
          for (int kx = 0; kx < WSZ; ++kx, ++c) {
            float v = Srow[c] + bias_table[by - kx * NH];
            Srow[c] = v;
            mx = fmaxf(mx, v);
          }
        }
      }
      // pass 2: exp + sum
      float sum = 0.f;
      for (int c = 0; c < NTOK; ++c) {
        float e = __expf(Srow[c] - mx);
        sum += e;
        Srow[c] = e;
      }
      // pass 3: normalize, convert to f16, vectorized b64 stores
      float inv = 1.0f / sum;
#pragma unroll
      for (int c4 = 0; c4 < 16; ++c4) {
        v4h pv;
#pragma unroll
        for (int i = 0; i < 4; ++i) {
          int c = c4 * 4 + i;
          pv[i] = (c < NTOK) ? (_Float16)(Srow[c] * inv) : (_Float16)0.f;
        }
        *(v4h*)&Prow[c4 * 4] = pv;
      }
    } else {
      v4h z = {};
#pragma unroll
      for (int c4 = 0; c4 < 16; ++c4) *(v4h*)&Prow[c4 * 4] = z;  // zero pad
    }
  }
  __syncthreads();

  // ---------------- Stage 5: O_h = P_h V_h  (one wave per head) -----------
  {
    int h = wv;
    const _Float16* Ph = &R1.P[h * 64 * P_STRIDE];
    const _Float16* Vh = &Vt[h * HD * VT_STRIDE];
    // Hoist all 4 V fragments (2 N-tiles x 2 K-steps) of this head.
    v16h vf[2][2];
#pragma unroll
    for (int nt = 0; nt < 2; ++nt)
#pragma unroll
      for (int kk = 0; kk < 2; ++kk)
        vf[nt][kk] =
            frag_load(Vh + nt * 16 * VT_STRIDE, lane, VT_STRIDE, kk * 32);
#pragma unroll
    for (int mt = 0; mt < 4; ++mt) {
      v16h a0 = frag_load(Ph + mt * 16 * P_STRIDE, lane, P_STRIDE, 0);
      v16h a1 = frag_load(Ph + mt * 16 * P_STRIDE, lane, P_STRIDE, 32);
#pragma unroll
      for (int nt = 0; nt < 2; ++nt) {
        v8f acc = {};
        acc = wmma_f16(a0, vf[nt][0], acc);
        acc = wmma_f16(a1, vf[nt][1], acc);
        int n = lane & 15, mb = (lane & 16) >> 1;
        _Float16* dst =
            &R0.Oh[(mt * 16 + mb) * XH_STRIDE + h * HD + nt * 16 + n];
#pragma unroll
        for (int j = 0; j < 8; ++j) dst[j * XH_STRIDE] = (_Float16)acc[j];
      }
    }
  }
  __syncthreads();

  // ---------------- Stage 6: projection GEMM + bias, scatter to out -------
  for (int mt = 0; mt < 4; ++mt) {
    // Hoist the 4 K-step A-fragments of this M tile (reused by 2 N-tiles).
    v16h a[4];
#pragma unroll
    for (int kk = 0; kk < 4; ++kk)
      a[kk] = frag_load(&R0.Oh[mt * 16 * XH_STRIDE], lane, XH_STRIDE, kk * 32);
    for (int nt = wv; nt < 8; nt += 4) {
      v8f acc = {};
#pragma unroll
      for (int kk = 0; kk < 4; ++kk) {
        v16h bf = frag_load(wprojT + nt * 16 * DIM, lane, DIM, kk * 32);
        acc = wmma_f16(a[kk], bf, acc);
      }
      int n = lane & 15, mb = (lane & 16) >> 1;
      int col = nt * 16 + n;
      float pb = proj_b[col];
#pragma unroll
      for (int j = 0; j < 8; ++j) {
        int row = mt * 16 + mb + j;
        if (row < NTOK) {
          float* dst =
              owin + (size_t)((row / WSZ) * 56 + (row % WSZ)) * DIM + col;
          *dst = acc[j] + pb;
        }
      }
    }
  }
}

// ---------------------------------------------------------------------------
extern "C" void kernel_launch(void* const* d_in, const int* in_sizes, int n_in,
                              void* d_out, int out_size, void* d_ws, size_t ws_size,
                              hipStream_t stream) {
  const float* x          = (const float*)d_in[0];
  const float* qkv_w      = (const float*)d_in[1];
  const float* qkv_b      = (const float*)d_in[2];
  const float* proj_w     = (const float*)d_in[3];
  const float* proj_b     = (const float*)d_in[4];
  const float* bias_table = (const float*)d_in[5];
  float* out = (float*)d_out;

  _Float16* wq = (_Float16*)d_ws;         // 384*128 f16
  _Float16* wp = wq + 384 * 128;          // 128*128 f16

  prep_weights<<<(384 * 128 + 128 * 128 + 255) / 256, 256, 0, stream>>>(
      qkv_w, proj_w, wq, wp);

  // 64 images * 8*8 windows = 4096 workgroups, 128 threads (4 wave32)
  swin_window_attention<<<4096, 128, 0, stream>>>(
      x, qkv_b, proj_b, bias_table, wq, wp, out);
}
